// MeanPooling_Aggregator_37254546326091
// MI455X (gfx1250) — matrix operand order repaired and checked by reading, not compiled, not run on hardware
//
#include <hip/hip_runtime.h>

typedef _Float16 v16h __attribute__((ext_vector_type(16)));
typedef float    v8f  __attribute__((ext_vector_type(8)));

__device__ __forceinline__ float frelu(float v) { return v > 0.0f ? v : 0.0f; }

// ---------------------------------------------------------------------------
// Generic [M,64] @ [64,64] GEMM on v_wmma_f32_16x16x32_f16.
//   A      : [M,64] f32 row-major
//   cnt    : optional per-row count; if non-null, A row is scaled by 1/max(cnt,1)
//            (fuses the segment-mean divide into the GEMM's A load)
//   W      : [64,64] f32 row-major (k-major), staged transposed into LDS as f16
//   bias   : optional [64] f32 added per output column
//   C      : output, row stride ldc (lets us write directly into the concat halves)
// Block = 256 threads = 8 waves; wave w computes rows [blk*128 + w*16, +16) x all 64 cols.
// All 256 threads stay active through the WMMAs (EXEC all-ones requirement);
// out-of-range rows are clamped on load and masked on store.
// ---------------------------------------------------------------------------
__global__ __launch_bounds__(256) void gemm64_wmma(
    const float* __restrict__ A, const float* __restrict__ cnt,
    const float* __restrict__ W, const float* __restrict__ bias,
    float* __restrict__ C, int M, int ldc, int do_relu)
{
    __shared__ _Float16 Wt[64 * 64];   // Wt[n][k] = W[k][n], 8 KB

    const int tid = threadIdx.x;
    for (int idx = tid; idx < 64 * 64; idx += 256) {
        const int k = idx >> 6, n = idx & 63;
        Wt[n * 64 + k] = (_Float16)W[idx];
    }
    __syncthreads();

    const int wave = tid >> 5;
    const int lane = tid & 31;
    const int hf   = lane >> 4;     // which half of the wave
    const int r    = lane & 15;
    const int m0   = (blockIdx.x * 8 + wave) << 4;

    const int row  = m0 + r;
    const int rowc = row < M ? row : M - 1;      // clamp so loads stay in-bounds

    float scale = 1.0f;
    if (cnt) {
        const float c = cnt[rowc];
        scale = 1.0f / (c > 1.0f ? c : 1.0f);
    }

    v8f acc[4] = {};   // 4 column tiles of 16

    #pragma unroll
    for (int s = 0; s < 2; ++s) {                // K split: k0 = 0, 32
        const int k0 = s * 32;

        // A fragment: lane<16 -> K {k0..k0+7, k0+16..k0+23}; lane>=16 -> +8
        const float* ap = A + (size_t)rowc * 64 + k0 + hf * 8;
        v16h a;
        #pragma unroll
        for (int i = 0; i < 8; ++i) {
            a[i]     = (_Float16)(ap[i]      * scale);
            a[i + 8] = (_Float16)(ap[i + 16] * scale);
        }

        #pragma unroll
        for (int t = 0; t < 4; ++t) {            // 4 N tiles of 16 columns
            // B fragment: lane holds col n0+r, K {k0+hf*8..+7, k0+16+hf*8..+7}
            const _Float16* bp = Wt + (size_t)(t * 16 + r) * 64 + k0 + hf * 8;
            v16h b;
            #pragma unroll
            for (int i = 0; i < 8; ++i) {
                b[i]     = bp[i];
                b[i + 8] = bp[i + 16];
            }
            acc[t] = __builtin_amdgcn_wmma_f32_16x16x32_f16(
                false, a, false, b, (short)0, acc[t], false, false);
        }
    }

    // C/D layout: element j of v8f -> row m0 + hf*8 + j, col t*16 + r
    #pragma unroll
    for (int t = 0; t < 4; ++t) {
        const int colv = t * 16 + r;
        const float bv = bias ? bias[colv] : 0.0f;
        #pragma unroll
        for (int j = 0; j < 8; ++j) {
            const int rr = m0 + hf * 8 + j;
            if (rr < M) {
                float v = acc[t][j] + bv;
                if (do_relu) v = frelu(v);
                C[(size_t)rr * ldc + colv] = v;
            }
        }
    }
}

// Zero a float buffer (seg_sum + cnt live contiguously in workspace).
__global__ void zero_f32(float* __restrict__ p, int n)
{
    const int i = blockIdx.x * blockDim.x + threadIdx.x;
    if (i < n) p[i] = 0.0f;
}

// ---------------------------------------------------------------------------
// Per-edge pass: seg[row] += relu(w * y[col] + mlp_bias), cnt[row] += 1.
// 8 threads per edge, 8 floats each (two float4 loads from the L2-resident y),
// accumulation via hardware global_atomic_add_f32 into the L2-resident seg.
// ---------------------------------------------------------------------------
__global__ __launch_bounds__(256) void edge_agg(
    const int* __restrict__ rowi, const int* __restrict__ coli,
    const float* __restrict__ ew, const float* __restrict__ y,
    const float* __restrict__ mb, float* __restrict__ seg,
    float* __restrict__ cnt, int E)
{
    const int gid = blockIdx.x * 256 + threadIdx.x;
    const int e = gid >> 3;
    if (e >= E) return;
    const int part = gid & 7;

    const int   rs = rowi[e];
    const int   cs = coli[e];
    const float w  = ew[e];

    const float4* yp = (const float4*)(y + (size_t)cs * 64 + part * 8);
    const float4  v0 = yp[0];
    const float4  v1 = yp[1];
    const float4* bp = (const float4*)(mb + part * 8);
    const float4  b0 = bp[0];
    const float4  b1 = bp[1];

    float h[8];
    h[0] = frelu(w * v0.x + b0.x);  h[1] = frelu(w * v0.y + b0.y);
    h[2] = frelu(w * v0.z + b0.z);  h[3] = frelu(w * v0.w + b0.w);
    h[4] = frelu(w * v1.x + b1.x);  h[5] = frelu(w * v1.y + b1.y);
    h[6] = frelu(w * v1.z + b1.z);  h[7] = frelu(w * v1.w + b1.w);

    float* dst = seg + (size_t)rs * 64 + part * 8;
    #pragma unroll
    for (int i = 0; i < 8; ++i) unsafeAtomicAdd(dst + i, h[i]);

    if (part == 0) unsafeAtomicAdd(cnt + rs, 1.0f);
}

extern "C" void kernel_launch(void* const* d_in, const int* in_sizes, int n_in,
                              void* d_out, int out_size, void* d_ws, size_t ws_size,
                              hipStream_t stream)
{
    const float* x      = (const float*)d_in[0];   // [N,64]
    const int*   ei     = (const int*)  d_in[1];   // [2,E]
    const float* ew     = (const float*)d_in[2];   // [E]
    const float* mlp_k  = (const float*)d_in[3];   // [64,64]
    const float* mlp_b  = (const float*)d_in[4];   // [64]
    const float* ngh_k  = (const float*)d_in[5];   // [64,64]
    const float* self_k = (const float*)d_in[6];   // [64,64]
    const float* bias   = (const float*)d_in[7];   // [128]
    float* out = (float*)d_out;                    // [N,128]

    const int N = in_sizes[0] / 64;
    const int E = in_sizes[1] / 2;

    // workspace: y[N*64] | seg[N*64] | cnt[N]   (seg & cnt contiguous -> one zero pass)
    float* y   = (float*)d_ws;
    float* seg = y + (size_t)N * 64;
    float* cnt = seg + (size_t)N * 64;

    const int* rowp = ei;        // edge_index[0]
    const int* colp = ei + E;    // edge_index[1]

    const int gblocks = (N + 127) / 128;

    // 1) y = x @ mlp_kernel  (no bias/relu: applied per-edge after scaling)
    gemm64_wmma<<<gblocks, 256, 0, stream>>>(x, nullptr, mlp_k, nullptr, y, N, 64, 0);

    // 2) zero seg_sum + cnt
    const int nz = N * 64 + N;
    zero_f32<<<(nz + 255) / 256, 256, 0, stream>>>(seg, nz);

    // 3) per-edge accumulate: seg[row] += relu(w * y[col] + mlp_bias); cnt[row] += 1
    edge_agg<<<(E * 8 + 255) / 256, 256, 0, stream>>>(rowp, colp, ew, y, mlp_b, seg, cnt, E);

    // 4) out[:, 0:64]  = relu((seg/max(cnt,1)) @ neighs_kernel + bias[0:64])
    gemm64_wmma<<<gblocks, 256, 0, stream>>>(seg, cnt, ngh_k, bias, out, N, 128, 1);

    // 5) out[:, 64:128] = relu(x @ self_kernel + bias[64:128])
    gemm64_wmma<<<gblocks, 256, 0, stream>>>(x, nullptr, self_k, bias + 64, out + 64, N, 128, 1);
}
